// AutoRegressive_52252572123273
// MI455X (gfx1250) — compile-verified, matrix operand order checked by reading, not run on hardware
//
#include <hip/hip_runtime.h>
#include <hip/hip_fp16.h>
#include <math.h>

typedef __attribute__((ext_vector_type(16))) _Float16 v16h;
typedef __attribute__((ext_vector_type(8)))  float    v8f;
typedef __attribute__((ext_vector_type(4)))  float    v4f;
typedef __attribute__((ext_vector_type(4)))  int      v4i;

#define NB    1024
#define TWU   256
#define TCD   1024
#define INF_  16
#define HIDN  32
#define OUTF  8

union HV { v16h h; v4i q[2]; };
union H8 { _Float16 h[8]; v4i q; };

// Branch-free activations on the TRANS pipe (v_exp_f32 / v_rcp_f32),
// co-executable with WMMA; exact at the +-inf limits.
__device__ __forceinline__ float sigm(float v) {
  return __builtin_amdgcn_rcpf(1.f + __expf(-v));
}
__device__ __forceinline__ float ftanh(float v) {
  return 1.f - 2.f * __builtin_amdgcn_rcpf(__expf(2.f * v) + 1.f);
}

__device__ __forceinline__ v8f wmma_acc(v16h a, v16h b, v8f c) {
  // D = A(16x32 f16) * B(32x16 f16) + C(16x16 f32)
  return __builtin_amdgcn_wmma_f32_16x16x32_f16(false, a, false, b, (short)0, c, false, false);
}
__device__ __forceinline__ v8f wmma_zc(v16h a, v16h b) {
  v8f z;                       // zero C: encodes as inline 0, no resident registers
#pragma unroll
  for (int r = 0; r < 8; ++r) z[r] = 0.f;
  return __builtin_amdgcn_wmma_f32_16x16x32_f16(false, a, false, b, (short)0, z, false, false);
}

__global__ __launch_bounds__(32, 1)
void lstm_ar_kernel(const float* __restrict__ x,   const float* __restrict__ ctx,
                    const float* __restrict__ Wih, const float* __restrict__ Whh,
                    const float* __restrict__ bih, const float* __restrict__ bhh,
                    const float* __restrict__ Wd,  const float* __restrict__ bd,
                    const int* __restrict__ lenx_g, const int* __restrict__ lenc_g,
                    float* __restrict__ out)
{
  // Acat row = 64 K-columns (h[0:32] | input[32:48] | zeros[48:64]) + 8 pad halves.
  // Row stride 144B: 16B aligned fragment loads, conflict-free across M rows.
  __shared__ alignas(16) _Float16 Acat[16][72];
  // Pre-swizzled W_ih B-fragments: [tile][half-group][lane][8 halves] (16B units).
  __shared__ alignas(16) _Float16 fB1[8][2][32][8];

  const int lane = threadIdx.x;     // 0..31, wave32
  const int n    = lane & 15;       // N column within a 16-wide tile
  const int hi   = lane >> 4;       // K-half selector per ISA fragment layout
  const int b0   = blockIdx.x * 16; // this wave's batch tile

  // ---------------- one-time: weight fragments ----------------
  v16h whh[8];            // W_hh^T chunk (recurrent), register resident
#pragma unroll
  for (int t = 0; t < 8; ++t) {
    const int row = t * 16 + n;     // gate row (PyTorch order i,f,g,o)
#pragma unroll
    for (int i = 0; i < 16; ++i) {
      const int K = ((i & 8) ? 16 : 0) + 8 * hi + (i & 7);
      whh[t][i] = (_Float16)Whh[row * HIDN + K];
      fB1[t][i >> 3][lane][i & 7] =
          (K < INF_) ? (_Float16)Wih[row * INF_ + K] : (_Float16)0.f;
    }
  }
  v16h wdy;               // W_d^T padded to N=16
#pragma unroll
  for (int i = 0; i < 16; ++i) {
    const int K = ((i & 8) ? 16 : 0) + 8 * hi + (i & 7);
    wdy[i] = (n < OUTF) ? (_Float16)Wd[n * HIDN + K] : (_Float16)0.f;
  }
  float bsum[8];          // per-tile gate bias (scalar per lane, added post-WMMA)
#pragma unroll
  for (int t = 0; t < 8; ++t) bsum[t] = bih[t * 16 + n] + bhh[t * 16 + n];
  const float ybias = (n < OUTF) ? bd[n] : 0.f;

  int lx[8], lc[8];
#pragma unroll
  for (int r = 0; r < 8; ++r) {
    lx[r] = lenx_g[b0 + r + 8 * hi];
    lc[r] = lenc_g[b0 + r + 8 * hi];
  }

  // zero Acat (h0 = 0, padding K 48..63 = 0)
  for (int idx = lane; idx < 16 * 72; idx += 32)
    (&Acat[0][0])[idx] = (_Float16)0.f;
  __syncthreads();

  v8f cst[2];             // cell state, lane-local per ISA C/D layout
#pragma unroll
  for (int jt = 0; jt < 2; ++jt)
#pragma unroll
    for (int r = 0; r < 8; ++r) cst[jt][r] = 0.f;

  auto loadA = [&](int c) -> v16h {          // A fragment for K-chunk c
    HV u;
    const _Float16* p = &Acat[n][32 * c + 8 * hi];
    u.q[0] = *(const v4i*)p;
    u.q[1] = *(const v4i*)(p + 16);
    return u.h;
  };
  auto loadB1 = [&](int t) -> v16h {         // W_ih fragment (LDS resident)
    HV u;
    u.q[0] = *(const v4i*)&fB1[t][0][lane][0];
    u.q[1] = *(const v4i*)&fB1[t][1][lane][0];
    return u.h;
  };

  // ---------------- warm-up: 256 masked LSTM steps ----------------
#pragma unroll 1
  for (int t = 0; t < TWU; ++t) {
    { // stage x_t -> Acat cols 32..47 (f32 -> f16), fully coalesced
      const int mm = lane >> 1, f0 = (lane & 1) * 8;
      const float* px = x + ((size_t)(b0 + mm) * TWU + t) * INF_ + f0;
      const v4f xa = *(const v4f*)px;
      const v4f xb = *(const v4f*)(px + 4);
      H8 hb;
#pragma unroll
      for (int j = 0; j < 4; ++j) { hb.h[j] = (_Float16)xa[j]; hb.h[4 + j] = (_Float16)xb[j]; }
      *(v4i*)&Acat[mm][32 + f0] = hb.q;
    }
    __syncthreads();

    const v16h a0 = loadA(0);                 // h chunk
    const v16h a1 = loadA(1);                 // [x | 0] chunk
    v8f acc[8];
#pragma unroll
    for (int g = 0; g < 8; ++g) {
      const v8f p = wmma_zc(a0, whh[g]);          // h @ W_hh^T
      acc[g] = wmma_acc(a1, loadB1(g), p);        // + x @ W_ih^T
    }

#pragma unroll
    for (int jt = 0; jt < 2; ++jt)
#pragma unroll
      for (int r = 0; r < 8; ++r) {
        const float zi = acc[0 + jt][r] + bsum[0 + jt];
        const float zf = acc[2 + jt][r] + bsum[2 + jt];
        const float zg = acc[4 + jt][r] + bsum[4 + jt];
        const float zo = acc[6 + jt][r] + bsum[6 + jt];
        const float co = cst[jt][r];
        const float cn = sigm(zf) * co + sigm(zi) * ftanh(zg);
        const float hn = sigm(zo) * ftanh(cn);
        const bool keep = t < lx[r];          // freeze state past length
        cst[jt][r] = keep ? cn : co;
        if (keep) Acat[r + 8 * hi][jt * 16 + n] = (_Float16)hn;
      }
    __syncthreads();
  }

  // ---------------- element = h @ W_d^T + b_d; out[:,0,:] ----------------
  v16h a0 = loadA(0);
  {
    const v8f ye = wmma_zc(a0, wdy);
    if (n < OUTF) {
#pragma unroll
      for (int r = 0; r < 8; ++r) {
        const int b = b0 + r + 8 * hi;
        const float v = ye[r] + ybias;
        out[((size_t)b * TCD) * OUTF + n] = (0 < lc[r]) ? v : 0.f;
        Acat[r + 8 * hi][32 + n] = (_Float16)v;   // element -> input cols 32..39
      }
    }
  }
  __syncthreads();

  // ---------------- decode: 1023 autoregressive steps ----------------
#pragma unroll 1
  for (int s = 0; s < TCD - 1; ++s) {
    if (lane < 16) { // stage ctx_s -> Acat cols 40..47
      const float* pc = ctx + ((size_t)(b0 + lane) * TCD + s) * (INF_ - OUTF);
      const v4f ca = *(const v4f*)pc;
      const v4f cb = *(const v4f*)(pc + 4);
      H8 hb;
#pragma unroll
      for (int j = 0; j < 4; ++j) { hb.h[j] = (_Float16)ca[j]; hb.h[4 + j] = (_Float16)cb[j]; }
      *(v4i*)&Acat[lane][40] = hb.q;
    }
    __syncthreads();

    const v16h a1 = loadA(1);                 // [element|ctx|0] chunk
    v8f acc[8];
#pragma unroll
    for (int g = 0; g < 8; ++g) {
      const v8f p = wmma_zc(a0, whh[g]);
      acc[g] = wmma_acc(a1, loadB1(g), p);
    }

#pragma unroll
    for (int jt = 0; jt < 2; ++jt)
#pragma unroll
      for (int r = 0; r < 8; ++r) {
        const float zi = acc[0 + jt][r] + bsum[0 + jt];
        const float zf = acc[2 + jt][r] + bsum[2 + jt];
        const float zg = acc[4 + jt][r] + bsum[4 + jt];
        const float zo = acc[6 + jt][r] + bsum[6 + jt];
        const float cn = sigm(zf) * cst[jt][r] + sigm(zi) * ftanh(zg);
        const float hn = sigm(zo) * ftanh(cn);
        cst[jt][r] = cn;
        Acat[r + 8 * hi][jt * 16 + n] = (_Float16)hn;
      }
    __syncthreads();

    a0 = loadA(0);                            // reload h_new; reused next step
    const v8f ye = wmma_zc(a0, wdy);          // y = h_new @ W_d^T (+ b_d below)
    if (n < OUTF) {
#pragma unroll
      for (int r = 0; r < 8; ++r) {
        const int b = b0 + r + 8 * hi;
        out[((size_t)b * TCD + (s + 1)) * OUTF + n] =
            ((s + 1) < lc[r]) ? (ye[r] + ybias) : 0.f;
      }
    }
  }
}

extern "C" void kernel_launch(void* const* d_in, const int* in_sizes, int n_in,
                              void* d_out, int out_size, void* d_ws, size_t ws_size,
                              hipStream_t stream) {
  (void)in_sizes; (void)n_in; (void)d_ws; (void)ws_size; (void)out_size;
  const float* x   = (const float*)d_in[0];
  const float* ctx = (const float*)d_in[1];
  const float* Wih = (const float*)d_in[2];
  const float* Whh = (const float*)d_in[3];
  const float* bih = (const float*)d_in[4];
  const float* bhh = (const float*)d_in[5];
  const float* Wd  = (const float*)d_in[6];
  const float* bd  = (const float*)d_in[7];
  const int* lx    = (const int*)d_in[8];
  const int* lc    = (const int*)d_in[9];
  float* out = (float*)d_out;

  lstm_ar_kernel<<<dim3(NB / 16), dim3(32), 0, stream>>>(
      x, ctx, Wih, Whh, bih, bhh, Wd, bd, lx, lc, out);
}
